// ColourCatDSSGINConv_41094247088190
// MI455X (gfx1250) — matrix operand order repaired and checked by reading, not compiled, not run on hardware
//
#include <hip/hip_runtime.h>
#include <hip/hip_bf16.h>

// ---------------------------------------------------------------------------
// ColourCatDSSGINConv for MI455X (gfx1250, wave32, WMMA)
//
// N=50000, E=800000, IN=16, CD=8, S=4, D=24, H=EMB=64.
// Working set (~150MB) fits the 192MB L2 -> edge scatter-add with
// global_atomic_add_f32 runs at L2 speed; GEMMs (2.5 GFLOP total) use
// v_wmma_f32_16x16x32_f16 with f32 accumulate, K padded 24->32.
// A-fragments: global_load_b128 (16B-aligned 8-float runs per lane).
// B-fragments: W staged transposed/f16/padded in LDS, read as ds_load_b128.
// Edge gather prefetched with global_prefetch_b8.
// ---------------------------------------------------------------------------

#define NN   50000
#define EE   800000
#define DD   24               // IN + CD
#define SS   4
#define HH   64               // hidden = emb
#define NSD  (NN * SS * DD)   // hf elements = N*96
#define MS   (NN * SS)        // 200000 rows, siamese MLP batch

typedef __attribute__((ext_vector_type(16))) _Float16 v16h;
typedef __attribute__((ext_vector_type(8)))  _Float16 v8h;
typedef __attribute__((ext_vector_type(8)))  float    v8f;

// Fragment maps (CDNA5 ISA 7.12.2, wave32):
//  A 16x32 f16 : lane(0-15) M=lane, K in {0..7,16..23}; lane(16-31) K += 8.
//    -> element i<8 : K = 8*laneHi + i ; i>=8 : K = 16 + 8*laneHi + (i-8)
//  B 32x16 f16 : lane N = lane&15, element i<8 : K = 8*laneHi + i ; i>=8 : K = 16 + 8*laneHi + (i-8)
//  C/D 16x16 f32: reg r -> row r + 8*laneHi, col = lane&15

// ---------------------------------------------------------------------------
// Kernel 1: hf[N, S*D] = concat(x, c); zero agg + BN-stat accumulators.
// ---------------------------------------------------------------------------
__global__ __launch_bounds__(256)
void prep_kernel(const float* __restrict__ x,   // [N,16]
                 const float* __restrict__ c,   // [N,S,8]
                 float* __restrict__ hf,        // [N,96]
                 float* __restrict__ agg,       // [N,96]
                 float* __restrict__ stats)     // 4*64 floats to zero
{
    int gid = blockIdx.x * blockDim.x + threadIdx.x;
    if (gid < 4 * HH) stats[gid] = 0.0f;
    if (gid >= NSD) return;
    int n = gid / (SS * DD);
    int j = gid - n * (SS * DD);
    int s = j / DD;
    int d = j - s * DD;
    float v = (d < 16) ? x[n * 16 + d] : c[(n * SS + s) * 8 + (d - 16)];
    hf[gid]  = v;
    agg[gid] = 0.0f;
}

// ---------------------------------------------------------------------------
// Kernel 2: agg[dst] += hf[src] (96 floats/edge). Two edges per wave; the
// second edge's source row is prefetched (global_prefetch_b8) while the first
// edge's gather+atomics are in flight.
// ---------------------------------------------------------------------------
__global__ __launch_bounds__(256)
void edge_scatter_kernel(const int* __restrict__ ei,   // [2,E]
                         const float* __restrict__ hf,
                         float* __restrict__ agg)
{
    int gw   = (blockIdx.x * blockDim.x + threadIdx.x) >> 5;  // global wave id
    int lane = threadIdx.x & 31;
    int e0   = gw * 2;                 // EE is even: e0 valid => e0+1 valid
    if (e0 >= EE) return;

    int src0 = ei[e0];
    int dst0 = ei[EE + e0];
    int src1 = ei[e0 + 1];
    int dst1 = ei[EE + e0 + 1];

    const float* hs1 = hf + (size_t)src1 * 96;
    __builtin_prefetch(hs1 + lane,      0, 0);   // 3 x 128B lines of next row
    __builtin_prefetch(hs1 + 32 + lane, 0, 0);
    __builtin_prefetch(hs1 + 64 + lane, 0, 0);

    const float* hs0 = hf + (size_t)src0 * 96;
    float*       ad0 = agg + (size_t)dst0 * 96;
#pragma unroll
    for (int j = 0; j < 3; ++j)
        atomicAdd(&ad0[lane + 32 * j], hs0[lane + 32 * j]);

    float* ad1 = agg + (size_t)dst1 * 96;
#pragma unroll
    for (int j = 0; j < 3; ++j)
        atomicAdd(&ad1[lane + 32 * j], hs1[lane + 32 * j]);
}

// helper: dst[0..3] += ep*h[0..3] + a[0..3]  (b128 loads)
__device__ __forceinline__ void acc4(float* dst, const float* __restrict__ h,
                                     const float* __restrict__ a, float ep) {
    float4 hv = *(const float4*)h;
    float4 av = *(const float4*)a;
    dst[0] += ep * hv.x + av.x;
    dst[1] += ep * hv.y + av.y;
    dst[2] += ep * hv.z + av.z;
    dst[3] += ep * hv.w + av.w;
}

// ---------------------------------------------------------------------------
// Kernel 3: GEMM1 + bias + BN statistics.
//  MODE 0: row r of 200000: m = (1+eps)*hf[r,:24] + agg[r,:24]
//  MODE 1: row n of 50000 : m = mean_s((1+eps)*h[n,s,:] + agg[n,s,:])
//  Z = m @ W1 + b1 ; per-column sum/sumsq accumulated atomically.
// W1 staged in LDS transposed [col][k], f16, K padded to 32, stride 40 halves.
// ---------------------------------------------------------------------------
template <int MODE>
__global__ __launch_bounds__(256)
void gemm1_bn_kernel(const float* __restrict__ hf,
                     const float* __restrict__ agg,
                     const float* __restrict__ epsp,   // scalar
                     const float* __restrict__ W1,     // [24,64]
                     const float* __restrict__ b1,     // [64]
                     float* __restrict__ Z,            // [M,64]
                     float* __restrict__ colsum,       // [64]
                     float* __restrict__ colsumsq,     // [64]
                     int Mrows)
{
    __shared__ __align__(16) _Float16 ldsW[HH * 40];  // [col][k], k<32 used

    // cooperative stage: transpose + f16 + zero-pad K 24->32
    for (int idx = threadIdx.x; idx < HH * 32; idx += 256) {
        int col = idx >> 5;
        int k   = idx & 31;
        ldsW[col * 40 + k] = (_Float16)((k < DD) ? W1[k * HH + col] : 0.0f);
    }
    __syncthreads();

    int wave   = threadIdx.x >> 5;
    int lane   = threadIdx.x & 31;
    int laneHi = lane >> 4;
    int nlane  = lane & 15;
    int mbase  = (blockIdx.x * 8 + wave) * 16;
    if (mbase >= Mrows) return;            // wave-uniform (Mrows % 16 == 0)

    float ep    = 1.0f + epsp[0];
    int   row   = mbase + nlane;           // A-fragment: M = lane & 15
    int   base0 = laneHi * 8;

    // A fragment: grp[0..7] -> K = base0..base0+7 ; grp[8..15] -> K = 16+base0..
    float grp[16];
#pragma unroll
    for (int i = 0; i < 16; ++i) grp[i] = 0.0f;

    if (MODE == 0) {
        const float* hr = hf  + (size_t)row * DD;
        const float* ar = agg + (size_t)row * DD;
        acc4(grp + 0, hr + base0,     ar + base0,     ep);
        acc4(grp + 4, hr + base0 + 4, ar + base0 + 4, ep);
        if (!laneHi) {                      // K = 16..23 valid; 24..31 stay 0
            acc4(grp + 8,  hr + 16, ar + 16, ep);
            acc4(grp + 12, hr + 20, ar + 20, ep);
        }
    } else {
#pragma unroll
        for (int s = 0; s < SS; ++s) {
            const float* hr = hf  + (size_t)row * (SS * DD) + s * DD;
            const float* ar = agg + (size_t)row * (SS * DD) + s * DD;
            acc4(grp + 0, hr + base0,     ar + base0,     ep);
            acc4(grp + 4, hr + base0 + 4, ar + base0 + 4, ep);
            if (!laneHi) {
                acc4(grp + 8,  hr + 16, ar + 16, ep);
                acc4(grp + 12, hr + 20, ar + 20, ep);
            }
        }
#pragma unroll
        for (int i = 0; i < 16; ++i) grp[i] *= 0.25f;
    }

    v16h afrag;
#pragma unroll
    for (int i = 0; i < 16; ++i) afrag[i] = (_Float16)grp[i];

#pragma unroll
    for (int t = 0; t < 4; ++t) {
        int n0  = t * 16;
        int col = n0 + nlane;

        // B fragment: element i<8 -> K = 8*laneHi + i ; i>=8 -> K = 16 + 8*laneHi + (i-8) ; two ds_load_b128
        const _Float16* wp = ldsW + col * 40 + laneHi * 8;
        v8h blo = *(const v8h*)(wp);
        v8h bhi = *(const v8h*)(wp + 16);
        v16h bfrag;
#pragma unroll
        for (int i = 0; i < 8; ++i) { bfrag[i] = blo[i]; bfrag[i + 8] = bhi[i]; }

        v8f cacc = {};
        cacc = __builtin_amdgcn_wmma_f32_16x16x32_f16(
            false, afrag, false, bfrag, (short)0, cacc, false, false);

        float bias = b1[col];
        float ps = 0.0f, ps2 = 0.0f;
#pragma unroll
        for (int r = 0; r < 8; ++r) {
            float z  = cacc[r] + bias;
            int   rr = mbase + r + (laneHi ? 8 : 0);
            Z[(size_t)rr * HH + col] = z;
            ps  += z;
            ps2 += z * z;
        }
        atomicAdd(&colsum[col],   ps);
        atomicAdd(&colsumsq[col], ps2);
    }
}

// ---------------------------------------------------------------------------
// Kernel 4: fold BN: scale = g*rsqrt(var+eps), shift = beta - mu*scale.
// ---------------------------------------------------------------------------
__global__ void bn_finalize_kernel(const float* __restrict__ colsum,
                                   const float* __restrict__ colsumsq,
                                   const float* __restrict__ g,
                                   const float* __restrict__ be,
                                   float invM,
                                   float* __restrict__ scale,
                                   float* __restrict__ shift)
{
    int i = threadIdx.x;
    if (i >= HH) return;
    float mu  = colsum[i] * invM;
    float var = colsumsq[i] * invM - mu * mu;
    float rs  = rsqrtf(var + 1e-5f);
    float sc  = g[i] * rs;
    scale[i]  = sc;
    shift[i]  = be[i] - mu * sc;
}

// ---------------------------------------------------------------------------
// Kernel 5: GEMM2, BN-affine + ReLU fused into the A load (b128 Z loads),
// K=64 as two chained WMMAs. W2 + scale/shift staged in LDS.
// ADD_SIAMESE fuses the DSS combine: out[N,S*EMB] row-major == [200000,64].
// ---------------------------------------------------------------------------
template <bool ADD_SIAMESE>
__global__ __launch_bounds__(256)
void gemm2_kernel(const float* __restrict__ Z,      // [M,64]
                  const float* __restrict__ scale,  // [64]
                  const float* __restrict__ shift,  // [64]
                  const float* __restrict__ W2,     // [64,64]
                  const float* __restrict__ b2,     // [64]
                  const float* __restrict__ ma,     // [N,64] or null
                  float* __restrict__ out,          // [M,64]
                  int Mrows)
{
    __shared__ __align__(16) _Float16 ldsW[HH * 72]; // [col][k], stride 72
    __shared__ float ldsScale[HH];
    __shared__ float ldsShift[HH];

    for (int idx = threadIdx.x; idx < HH * HH; idx += 256) {
        int col = idx >> 6;
        int k   = idx & 63;
        ldsW[col * 72 + k] = (_Float16)W2[k * HH + col];
    }
    if (threadIdx.x < HH) {
        ldsScale[threadIdx.x] = scale[threadIdx.x];
        ldsShift[threadIdx.x] = shift[threadIdx.x];
    }
    __syncthreads();

    int wave   = threadIdx.x >> 5;
    int lane   = threadIdx.x & 31;
    int laneHi = lane >> 4;
    int nlane  = lane & 15;
    int mbase  = (blockIdx.x * 8 + wave) * 16;
    if (mbase >= Mrows) return;            // wave-uniform exit

    int row   = mbase + nlane;
    int base0 = laneHi * 8;
    const float* zr = Z + (size_t)row * HH;

    // A fragments: group g covers K = 16*g + base0 .. +7 (b128 pairs),
    // with act = relu(z*scale + shift); g0,g1 -> a0, g2,g3 -> a1.
    v16h a0, a1;
#pragma unroll
    for (int g = 0; g < 4; ++g) {
        int off = 16 * g + base0;
        float4 z0 = *(const float4*)(zr + off);
        float4 z1 = *(const float4*)(zr + off + 4);
        float zz[8] = { z0.x, z0.y, z0.z, z0.w, z1.x, z1.y, z1.z, z1.w };
#pragma unroll
        for (int j = 0; j < 8; ++j) {
            int   k   = off + j;
            float act = fmaxf(zz[j] * ldsScale[k] + ldsShift[k], 0.0f);
            if (g < 2) a0[(g & 1) * 8 + j] = (_Float16)act;
            else       a1[(g & 1) * 8 + j] = (_Float16)act;
        }
    }

#pragma unroll
    for (int t = 0; t < 4; ++t) {
        int n0  = t * 16;
        int col = n0 + nlane;

        const _Float16* wp = ldsW + col * 72 + laneHi * 8;
        v8h p0 = *(const v8h*)(wp);            // K-step 0: K = 8*laneHi + 0..7, then 16 + 8*laneHi + 0..7
        v8h p1 = *(const v8h*)(wp + 16);
        v8h p2 = *(const v8h*)(wp + 32);       // K-step 1: +32
        v8h p3 = *(const v8h*)(wp + 48);
        v16h bf0, bf1;
#pragma unroll
        for (int i = 0; i < 8; ++i) {
            bf0[i] = p0[i]; bf0[i + 8] = p1[i];
            bf1[i] = p2[i]; bf1[i + 8] = p3[i];
        }

        v8f cacc = {};
        cacc = __builtin_amdgcn_wmma_f32_16x16x32_f16(
            false, a0, false, bf0, (short)0, cacc, false, false);
        cacc = __builtin_amdgcn_wmma_f32_16x16x32_f16(
            false, a1, false, bf1, (short)0, cacc, false, false);

        float bias = b2[col];
#pragma unroll
        for (int r = 0; r < 8; ++r) {
            int   rr = mbase + r + (laneHi ? 8 : 0);
            float y  = cacc[r] + bias;
            if (ADD_SIAMESE) y += ma[(size_t)(rr >> 2) * HH + col];
            out[(size_t)rr * HH + col] = y;
        }
    }
}

// ---------------------------------------------------------------------------
// Host launcher
// ---------------------------------------------------------------------------
extern "C" void kernel_launch(void* const* d_in, const int* in_sizes, int n_in,
                              void* d_out, int out_size, void* d_ws, size_t ws_size,
                              hipStream_t stream) {
    (void)in_sizes; (void)n_in; (void)out_size; (void)ws_size;

    const float* x    = (const float*)d_in[0];
    const float* c    = (const float*)d_in[1];
    const int*   ei   = (const int*)  d_in[2];
    const float* epsS = (const float*)d_in[3];
    const float* W1s  = (const float*)d_in[4];
    const float* b1s  = (const float*)d_in[5];
    const float* g1s  = (const float*)d_in[6];
    const float* be1s = (const float*)d_in[7];
    const float* W2s  = (const float*)d_in[8];
    const float* b2s  = (const float*)d_in[9];
    const float* epsA = (const float*)d_in[10];
    const float* W1a  = (const float*)d_in[11];
    const float* b1a  = (const float*)d_in[12];
    const float* g1a  = (const float*)d_in[13];
    const float* be1a = (const float*)d_in[14];
    const float* W2a  = (const float*)d_in[15];
    const float* b2a  = (const float*)d_in[16];
    float* out = (float*)d_out;                 // [N, S*EMB] == [200000, 64]

    // Workspace carve-up (floats); total ~115 MB, all offsets 16B-aligned.
    float* ws    = (float*)d_ws;
    float* hf    = ws;                          // N*96
    float* agg   = hf  + (size_t)NSD;           // N*96
    float* Z1    = agg + (size_t)NSD;           // 200000*64
    float* Za    = Z1  + (size_t)MS * HH;       // 50000*64
    float* maOut = Za  + (size_t)NN * HH;       // 50000*64
    float* stats = maOut + (size_t)NN * HH;     // 8*64 floats
    float* sumS  = stats;            float* sqS  = stats + HH;
    float* sumA  = stats + 2 * HH;   float* sqA  = stats + 3 * HH;
    float* sclS  = stats + 4 * HH;   float* shfS = stats + 5 * HH;
    float* sclA  = stats + 6 * HH;   float* shfA = stats + 7 * HH;

    // 1) build hf, zero agg + stats
    prep_kernel<<<(NSD + 255) / 256, 256, 0, stream>>>(x, c, hf, agg, stats);

    // 2) edge scatter-add (2 edges per wave, prefetch-ahead)
    {
        long threads = (long)(EE / 2) * 32;
        edge_scatter_kernel<<<(int)((threads + 255) / 256), 256, 0, stream>>>(ei, hf, agg);
    }

    // 3) GEMM1 + BN stats, both paths
    gemm1_bn_kernel<0><<<(MS + 127) / 128, 256, 0, stream>>>(
        hf, agg, epsS, W1s, b1s, Z1, sumS, sqS, MS);
    gemm1_bn_kernel<1><<<(NN + 127) / 128, 256, 0, stream>>>(
        hf, agg, epsA, W1a, b1a, Za, sumA, sqA, NN);

    // 4) BN fold
    bn_finalize_kernel<<<1, 64, 0, stream>>>(sumS, sqS, g1s, be1s, 1.0f / (float)MS, sclS, shfS);
    bn_finalize_kernel<<<1, 64, 0, stream>>>(sumA, sqA, g1a, be1a, 1.0f / (float)NN, sclA, shfA);

    // 5) GEMM2: aggregated path, then siamese path fused with DSS combine
    gemm2_kernel<false><<<(NN + 127) / 128, 256, 0, stream>>>(
        Za, sclA, shfA, W2a, b2a, nullptr, maOut, NN);
    gemm2_kernel<true><<<(MS + 127) / 128, 256, 0, stream>>>(
        Z1, sclS, shfS, W2s, b2s, maOut, out, MS);
}
